// FBSparseSparsityInvVGG_33809982554779
// MI455X (gfx1250) — matrix operand order repaired and checked by reading, not compiled, not run on hardware
//
#include <hip/hip_runtime.h>

#define BATCH 32
#define NCLS  101

typedef __attribute__((ext_vector_type(16))) __bf16 v16bf;
typedef __attribute__((ext_vector_type(8)))  float  v8f;

// ---------- bf16 helpers (round-to-nearest-even) ----------
__device__ __forceinline__ unsigned short f2bf(float f) {
    union { float f; unsigned int u; } t; t.f = f;
    unsigned int r = t.u + 0x7FFFu + ((t.u >> 16) & 1u);
    return (unsigned short)(r >> 16);
}
__device__ __forceinline__ float bf2f(unsigned short h) {
    union { unsigned int u; float f; } t; t.u = ((unsigned int)h) << 16;
    return t.f;
}

// ---------- ingest: fp32 NCHW [B,2,191,255] -> bf16 NHWC (C padded to 16) ----------
__global__ void ingest_kernel(const float* __restrict__ x, unsigned short* __restrict__ act) {
    const int H = 191, W = 255;
    int w = blockIdx.x * blockDim.x + threadIdx.x;
    int h = blockIdx.y, b = blockIdx.z;
    if (w >= W) return;
    size_t p = ((size_t)b * H + h) * W + w;
    unsigned short* q = act + p * 16;
    size_t hw = (size_t)h * W + w;
#pragma unroll
    for (int c = 0; c < 16; ++c) {
        float v = (c < 2) ? x[((size_t)b * 2 + c) * (H * W) + hw] : 0.0f;
        q[c] = f2bf(v);
    }
}

// ---------- mask from values: active if sum_c |x| > 0 ----------
__global__ void mask_kernel(const unsigned short* __restrict__ act,
                            int H, int W, int C, float* __restrict__ mask) {
    int w = blockIdx.x * blockDim.x + threadIdx.x;
    int h = blockIdx.y, b = blockIdx.z;
    if (w >= W) return;
    size_t p = ((size_t)b * H + h) * W + w;
    const unsigned short* px = act + p * C;
    float s = 0.0f;
    for (int c = 0; c < C; ++c) s += fabsf(bf2f(px[c]));
    mask[p] = (s > 0.0f) ? 1.0f : 0.0f;
}

// ---------- normalize_input: x / (conv3x3(mask, ones_w) + 1e-5) ----------
__global__ void norm_kernel(const unsigned short* __restrict__ act,
                            const float* __restrict__ mask,
                            const float* __restrict__ onesw,   // 9 taps (cross-correlation)
                            int H, int W, int C,
                            unsigned short* __restrict__ xn) {
    int w = blockIdx.x * blockDim.x + threadIdx.x;
    int h = blockIdx.y, b = blockIdx.z;
    if (w >= W) return;
    size_t p = ((size_t)b * H + h) * W + w;
    float denom = 1e-5f;
#pragma unroll
    for (int ky = 0; ky < 3; ++ky)
#pragma unroll
        for (int kx = 0; kx < 3; ++kx) {
            int ih = h + ky - 1, iw = w + kx - 1;
            if (ih >= 0 && ih < H && iw >= 0 && iw < W)
                denom += onesw[ky * 3 + kx] * mask[((size_t)b * H + ih) * W + iw];
        }
    float inv = 1.0f / denom;
    const unsigned short* px = act + p * C;
    unsigned short* qx = xn + p * C;
    for (int c = 0; c < C; ++c) qx[c] = f2bf(bf2f(px[c]) * inv);
}

// ---------- pack fp32 OIHW weights into WMMA-fragment-ready bf16 ----------
// Fragment dword j of lane = bf16 pair for K = ktile*32 + 16*(j>>2) + 8*(lane>>4) + 2*(j&3) (+0,+1),
// column = octile*16 + lane%16.  K layout: k = kpos * Cin + c  (kpos = ky*3+kx).
// grid = (Kpad/32, Co/16), block = 256 (one fragment tile per block)
__global__ void packw_frag_kernel(const float* __restrict__ w, int ci_real, int Cin, int cshift,
                                  int Co, unsigned int* __restrict__ Wf) {
    int ktile = blockIdx.x, octile = blockIdx.y;
    int tid = threadIdx.x;
    int j = tid & 7, lane = tid >> 3;
    int half = lane >> 4, n = lane & 15;
    int kb = (ktile << 5) + ((j >> 2) << 4) + (half << 3) + ((j & 3) << 1);
    int oc = (octile << 4) + n;
    unsigned int u = 0;
#pragma unroll
    for (int s = 0; s < 2; ++s) {
        int k = kb + s;
        float v = 0.0f;
        if (k < 9 * Cin) {
            int kpos = k >> cshift;
            int c = k & (Cin - 1);
            if (c < ci_real) v = w[((size_t)oc * ci_real + c) * 9 + kpos];
        }
        u |= ((unsigned int)f2bf(v)) << (16 * s);
    }
    Wf[((size_t)ktile * gridDim.y + octile) * 256 + tid] = u;
}

// ---------- implicit-GEMM conv: one wave = 16(ow) x (16*NT)oc, WMMA bf16->f32 ----------
// grid = (ceil(OW/16), OH, BATCH * (Co/(16*NT))); division-free indexing.
// Fused BN(eval)+ReLU+mask epilogue, bf16 NHWC out.
template <int NT>
__global__ __launch_bounds__(32)
void conv_wmma_kernel(const unsigned short* __restrict__ xn, int Hin, int Win, int Cin,
                      const unsigned int* __restrict__ Wf, int Co,
                      int OH, int OW, int stride, int pad,
                      const float* __restrict__ mask, int maskWindowed,
                      const float* __restrict__ bg, const float* __restrict__ bb,
                      const float* __restrict__ bm, const float* __restrict__ bv,
                      unsigned short* __restrict__ out) {
    const int lane = threadIdx.x;            // wave32: one wave per block
    const int half = lane >> 4;
    const int mrow = lane & 15;
    const int ntiles_oc = Co >> 4;

    const int b       = blockIdx.z & (BATCH - 1);
    const int octile0 = (int)(blockIdx.z >> 5) * NT;
    const int oh      = blockIdx.y;
    const int ow0     = blockIdx.x << 4;
    const int ow      = ow0 + mrow;

    const uint4* x4 = (const uint4*)xn;
    const uint4* w4 = (const uint4*)Wf;

    v8f acc[NT];
#pragma unroll
    for (int u = 0; u < NT; ++u) acc[u] = (v8f){0.f, 0.f, 0.f, 0.f, 0.f, 0.f, 0.f, 0.f};

    if (Cin == 16) {
        // K = 144 padded to 160 (5 K-tiles); fragment group g of tile t = tap kpos = 2t+g
#pragma unroll
        for (int t = 0; t < 5; ++t) {
            union { v16bf v; uint4 q[2]; } A;
#pragma unroll
            for (int g = 0; g < 2; ++g) {
                const int kpos = 2 * t + g;
                uint4 a = {0u, 0u, 0u, 0u};
                if (kpos < 9) {
                    const int ky = (kpos >= 6) ? 2 : ((kpos >= 3) ? 1 : 0);
                    const int kx = kpos - ky * 3;
                    int ih = oh * stride + ky - pad;
                    int iw = ow * stride + kx - pad;
                    if (ih >= 0 && ih < Hin && iw >= 0 && iw < Win) {
                        size_t e = (((size_t)(b * Hin + ih) * Win + iw) << 4) + 8 * half;
                        a = x4[e >> 3];
                    }
                }
                A.q[g] = a;
            }
            const uint4* bp = w4 + ((size_t)t * ntiles_oc + octile0) * 64 + lane * 2;
#pragma unroll
            for (int u = 0; u < NT; ++u) {
                union { v16bf v; uint4 q[2]; } Bf;
                Bf.q[0] = bp[(size_t)u * 64 + 0];
                Bf.q[1] = bp[(size_t)u * 64 + 1];
                acc[u] = __builtin_amdgcn_wmma_f32_16x16x32_bf16(
                    false, A.v, false, Bf.v, (short)0, acc[u], false, false);
            }
        }
    } else {
        // Cin >= 32 (multiple of 32): each 32-K step lies inside one tap
        const int csteps = Cin >> 5;
#pragma unroll
        for (int kpos = 0; kpos < 9; ++kpos) {
            const int ky = kpos / 3, kx = kpos % 3;         // constants after unroll
            int ih = oh * stride + ky - pad;
            int iw = ow * stride + kx - pad;
            bool ok = (ih >= 0) && (ih < Hin) && (iw >= 0) && (iw < Win);
            size_t eb = 0;
            if (ok) eb = ((size_t)(b * Hin + ih) * Win + iw) * Cin + 8 * half;
            int ktile = kpos * csteps;
            if (kpos < 8)   // prefetch next tap's first weight fragment
                __builtin_prefetch(w4 + ((size_t)(kpos + 1) * csteps * ntiles_oc + octile0) * 64 +
                                       lane * 2, 0, 1);
            for (int cc = 0; cc < Cin; cc += 32, ++ktile) {
                union { v16bf v; uint4 q[2]; } A;
                uint4 a0 = {0u, 0u, 0u, 0u}, a1 = {0u, 0u, 0u, 0u};
                if (ok) {
                    a0 = x4[(eb + cc) >> 3];
                    a1 = x4[(eb + cc + 16) >> 3];
                }
                A.q[0] = a0; A.q[1] = a1;
                const uint4* bp = w4 + ((size_t)ktile * ntiles_oc + octile0) * 64 + lane * 2;
#pragma unroll
                for (int u = 0; u < NT; ++u) {
                    union { v16bf v; uint4 q[2]; } Bf;
                    Bf.q[0] = bp[(size_t)u * 64 + 0];
                    Bf.q[1] = bp[(size_t)u * 64 + 1];
                    acc[u] = __builtin_amdgcn_wmma_f32_16x16x32_bf16(
                        false, A.v, false, Bf.v, (short)0, acc[u], false, false);
                }
            }
        }
    }

    // ---- epilogue: D layout VGPR r -> output column ow0 + r + 8*half, channel = tile*16 + lane%16
    float sc[NT], mn[NT], bt[NT];
#pragma unroll
    for (int u = 0; u < NT; ++u) {
        int oc = (octile0 + u) * 16 + mrow;
        sc[u] = bg[oc] * rsqrtf(bv[oc] + 1e-4f);
        mn[u] = bm[oc];
        bt[u] = bb[oc];
    }
    size_t ob0 = ((size_t)b * OH + oh) * OW;          // scalar
#pragma unroll
    for (int r = 0; r < 8; ++r) {
        int o2w = ow0 + r + 8 * half;
        if (o2w >= OW) continue;
        float mval;
        if (maskWindowed) {
            float mm = 0.0f;
#pragma unroll
            for (int ky = 0; ky < 3; ++ky)
#pragma unroll
                for (int kx = 0; kx < 3; ++kx) {
                    int ih = oh * stride + ky - pad, iw = o2w * stride + kx - pad;
                    if (ih >= 0 && ih < Hin && iw >= 0 && iw < Win)
                        mm = fmaxf(mm, mask[((size_t)b * Hin + ih) * Win + iw]);
                }
            mval = mm;
        } else {
            mval = mask[ob0 + o2w];
        }
        size_t obase = (ob0 + o2w) * Co;
#pragma unroll
        for (int u = 0; u < NT; ++u) {
            int oc = (octile0 + u) * 16 + mrow;
            float v = (acc[u][r] - mn[u]) * sc[u] + bt[u];
            v = fmaxf(v, 0.0f);
            v = (mval > 0.0f) ? v : 0.0f;
            out[obase + oc] = f2bf(v);
        }
    }
}

// ---------- sparse maxpool 3x3 stride 2 VALID ----------
__global__ void pool_kernel(const unsigned short* __restrict__ act,
                            const float* __restrict__ mask,
                            int H, int W, int C, unsigned short* __restrict__ out) {
    int OH = (H - 3) / 2 + 1, OW = (W - 3) / 2 + 1;
    int ow = blockIdx.x * blockDim.x + threadIdx.x;
    int oh = blockIdx.y, b = blockIdx.z;
    if (ow >= OW) return;
    size_t base[9]; int a9[9]; int nact = 0;
#pragma unroll
    for (int ky = 0; ky < 3; ++ky)
#pragma unroll
        for (int kx = 0; kx < 3; ++kx) {
            int ih = 2 * oh + ky, iw = 2 * ow + kx;          // VALID: always in bounds
            size_t pi = ((size_t)b * H + ih) * W + iw;
            int a = (mask[pi] > 0.0f);
            base[ky * 3 + kx] = pi * (size_t)C;
            a9[ky * 3 + kx] = a;
            nact += a;
        }
    unsigned short* q = out + (((size_t)b * OH + oh) * OW + ow) * C;
    for (int c = 0; c < C; ++c) {
        float best = -3.0e38f;
#pragma unroll
        for (int t = 0; t < 9; ++t)
            if (a9[t]) best = fmaxf(best, bf2f(act[base[t] + c]));
        q[c] = nact ? f2bf(best) : (unsigned short)0;
    }
}

// ---------- FC: [B, 256*2*3] (NCHW flatten) @ fc_w^T + fc_b ----------
__global__ void fc_kernel(const unsigned short* __restrict__ act,   // bf16 NHWC [B,2,3,256]
                          const float* __restrict__ fw, const float* __restrict__ fb,
                          float* __restrict__ out) {
    int cls = blockIdx.x * blockDim.x + threadIdx.x;
    int b = blockIdx.y;
    if (cls >= NCLS) return;
    float s = fb[cls];
    for (int c = 0; c < 256; ++c) {
#pragma unroll
        for (int hw = 0; hw < 6; ++hw) {
            int h = hw / 3, w = hw - h * 3;
            float v = bf2f(act[(((size_t)b * 2 + h) * 3 + w) * 256 + c]);
            s += v * fw[(size_t)cls * 1536 + c * 6 + hw];   // k = c*6 + h*3 + w
        }
    }
    out[(size_t)b * NCLS + cls] = s;
}

static inline int cdiv(int a, int b) { return (a + b - 1) / b; }
static inline int ilog2(int v) { int s = 0; while ((1 << s) < v) ++s; return s; }

static void launch_conv(int NT, hipStream_t stream,
                        const unsigned short* xn, int H, int W, int Cin,
                        const unsigned int* Wf, int Co, int OH, int OW, int stride, int pad,
                        const float* M, int maskWin,
                        const float* bg, const float* bb, const float* bm, const float* bv,
                        unsigned short* out) {
    dim3 g(cdiv(OW, 16), OH, BATCH * (Co / (16 * NT)));
    if (NT == 1)
        conv_wmma_kernel<1><<<g, 32, 0, stream>>>(xn, H, W, Cin, Wf, Co, OH, OW, stride, pad,
                                                  M, maskWin, bg, bb, bm, bv, out);
    else if (NT == 2)
        conv_wmma_kernel<2><<<g, 32, 0, stream>>>(xn, H, W, Cin, Wf, Co, OH, OW, stride, pad,
                                                  M, maskWin, bg, bb, bm, bv, out);
    else
        conv_wmma_kernel<4><<<g, 32, 0, stream>>>(xn, H, W, Cin, Wf, Co, OH, OW, stride, pad,
                                                  M, maskWin, bg, bb, bm, bv, out);
}

extern "C" void kernel_launch(void* const* d_in, const int* in_sizes, int n_in,
                              void* d_out, int out_size, void* d_ws, size_t ws_size,
                              hipStream_t stream) {
    (void)in_sizes; (void)n_in; (void)out_size; (void)ws_size;

    // Input order: 0:x | 1..12:conv_ws | 13..24:bn_g | 25..36:bn_b | 37..48:bn_m | 49..60:bn_v
    // 61:ones_ws [11,1,1,3,3] | 62:fc_w | 63:fc_b
    const float* x = (const float*)d_in[0];
    const float* convw[12]; for (int i = 0; i < 12; ++i) convw[i] = (const float*)d_in[1 + i];
    const float* bng[12];   for (int i = 0; i < 12; ++i) bng[i]   = (const float*)d_in[13 + i];
    const float* bnb[12];   for (int i = 0; i < 12; ++i) bnb[i]   = (const float*)d_in[25 + i];
    const float* bnm[12];   for (int i = 0; i < 12; ++i) bnm[i]   = (const float*)d_in[37 + i];
    const float* bnv[12];   for (int i = 0; i < 12; ++i) bnv[i]   = (const float*)d_in[49 + i];
    const float* onesw = (const float*)d_in[61];
    const float* fcw   = (const float*)d_in[62];
    const float* fcb   = (const float*)d_in[63];

    // Workspace (peak activation: 32*191*255 px * 16ch * 2B = 47.6 MiB)
    const size_t MiB = 1024ull * 1024ull;
    char* ws = (char*)d_ws;
    unsigned short* P0 = (unsigned short*)(ws);
    unsigned short* P1 = (unsigned short*)(ws + 48 * MiB);
    unsigned short* XN = (unsigned short*)(ws + 96 * MiB);
    float*          M  = (float*)(ws + 144 * MiB);
    unsigned int*   WF = (unsigned int*)(ws + 152 * MiB);

    const int Hs[11]  = {191, 191, 95, 95, 47, 47, 23, 23, 11, 11, 5};
    const int Wd[11]  = {255, 255, 127, 127, 63, 63, 31, 31, 15, 15, 7};
    const int cir[11] = {2, 16, 16, 32, 32, 64, 64, 128, 128, 256, 256};
    const int cps[11] = {16, 16, 16, 32, 32, 64, 64, 128, 128, 256, 256};
    const int cos_[11]= {16, 16, 32, 32, 64, 64, 128, 128, 256, 256, 512};
    const int poolAfter[11] = {0, 1, 0, 1, 0, 1, 0, 1, 0, 1, 0};

    ingest_kernel<<<dim3(cdiv(255, 64), 191, BATCH), 64, 0, stream>>>(x, P0);

    unsigned short* cur = P0;
    unsigned short* nxt = P1;

    for (int i = 0; i < 11; ++i) {
        const int H = Hs[i], W = Wd[i], Cin = cps[i], Co = cos_[i];
        const int K = 9 * Cin;
        const int Kpad = (K + 31) & ~31;
        const int NT = (Co >= 64) ? 4 : (Co >= 32 ? 2 : 1);

        dim3 ge(cdiv(W, 64), H, BATCH);
        mask_kernel<<<ge, 64, 0, stream>>>(cur, H, W, Cin, M);
        norm_kernel<<<ge, 64, 0, stream>>>(cur, M, onesw + i * 9, H, W, Cin, XN);
        packw_frag_kernel<<<dim3(Kpad >> 5, Co >> 4), 256, 0, stream>>>(
            convw[i], cir[i], Cin, ilog2(Cin), Co, WF);

        launch_conv(NT, stream, XN, H, W, Cin, WF, Co,
                    H, W, 1, 1, M, 0, bng[i], bnb[i], bnm[i], bnv[i], nxt);
        { unsigned short* t = cur; cur = nxt; nxt = t; }

        if (poolAfter[i]) {
            int OH = (H - 3) / 2 + 1, OW = (W - 3) / 2 + 1;
            pool_kernel<<<dim3(cdiv(OW, 64), OH, BATCH), 64, 0, stream>>>(cur, M, H, W, Co, nxt);
            unsigned short* t = cur; cur = nxt; nxt = t;
        }
    }

    // Final conv: 512->256, 3x3, stride 2, VALID; 5x7 -> 2x3. M holds the i=10 mask (5x7).
    {
        const int K = 9 * 512;                 // 4608, multiple of 32
        packw_frag_kernel<<<dim3(K >> 5, 256 >> 4), 256, 0, stream>>>(
            convw[11], 512, 512, ilog2(512), 256, WF);
        launch_conv(4, stream, cur, 5, 7, 512, WF, 256,
                    2, 3, 2, 0, M, 1, bng[11], bnb[11], bnm[11], bnv[11], nxt);
    }

    fc_kernel<<<dim3(cdiv(NCLS, 64), BATCH), 64, 0, stream>>>(nxt, fcw, fcb, (float*)d_out);
}
